// SparseAttentionModel_62629213110485
// MI455X (gfx1250) — matrix operand order repaired and checked by reading, not compiled, not run on hardware
//
#include <hip/hip_runtime.h>
#include <hip/hip_bf16.h>
#include <cstdint>
#include <cstddef>

typedef __attribute__((ext_vector_type(16))) _Float16 v16h;
typedef __attribute__((ext_vector_type(8)))  _Float16 v8h;
typedef __attribute__((ext_vector_type(8)))  float    v8f;
typedef __attribute__((ext_vector_type(4)))  int      v4i;

#if __has_builtin(__builtin_amdgcn_global_load_async_to_lds_b128) && \
    __has_builtin(__builtin_amdgcn_s_wait_asynccnt)
#define HAVE_ASYNC_LDS 1
typedef __attribute__((address_space(1))) v4i glb_v4i;
typedef __attribute__((address_space(3))) v4i lds_v4i;
#else
#define HAVE_ASYNC_LDS 0
#endif

namespace {
constexpr int kB  = 32;
constexpr int kN  = 512;
constexpr int kD  = 64;
constexpr int kH  = 6;
constexpr int kDH = 32;
constexpr int kCB = 16;
constexpr int kSB = 8;
constexpr int kOUT = 7;
constexpr int kHID = 32;
constexpr int knC = kN / kCB;       // 32
constexpr int knF = kN / kSB;       // 64
constexpr int kCD = kCB * kDH;      // 512
constexpr int kM  = kB * kN;        // 16384 token rows
constexpr int kQKVN = 3 * kH * kDH; // 576
constexpr int kMC = kB * kH * knC;  // 6144 chunk rows
constexpr float kSCALE = 0.17677669529663687f; // 32^-0.5
constexpr float kNEG   = -3.4e38f;
}

// ---------------------------------------------------------------------------
// WMMA fragment helpers (gfx1250 wave32, V_WMMA_F32_16X16X32_F16)
// A 16x32 f16, row-major. ISA 7.12.2: lanes 0-15 hold row r, halfs
// {K0..7,K16..23}; lanes 16-31 hold row r, halfs {K8..15,K24..31}
// => two contiguous 16B runs per lane.
// ---------------------------------------------------------------------------
__device__ __forceinline__ v16h load_a_frag(const _Float16* __restrict__ A,
                                            int lda, int row0, int k0, int lane) {
  const int hh = lane >> 4;
  const int r  = lane & 15;
  const _Float16* p = A + (size_t)(row0 + r) * lda + k0 + hh * 8;
  v8h lo = *(const v8h*)(p);
  v8h hi = *(const v8h*)(p + 16);
  v16h a;
#pragma unroll
  for (int i = 0; i < 8; ++i) { a[i] = lo[i]; a[i + 8] = hi[i]; }
  return a;
}

// Same fragment shape but sourced from an LDS-staged slab (ds_load_b128 x2).
__device__ __forceinline__ v16h load_a_frag_lds(const _Float16* sA, int lda,
                                                int k0, int lane) {
  const int hh = lane >> 4;
  const int r  = lane & 15;
  const _Float16* p = sA + r * lda + k0 + hh * 8;
  v8h lo = *(const v8h*)(p);
  v8h hi = *(const v8h*)(p + 16);
  v16h a;
#pragma unroll
  for (int i = 0; i < 8; ++i) { a[i] = lo[i]; a[i + 8] = hi[i]; }
  return a;
}

// B fragments come from a repacked weight buffer: tile (kt,nt) stores
// 32 lanes x 16 contiguous halfs, already in the hardware lane layout.
__device__ __forceinline__ v16h load_b_frag(const _Float16* __restrict__ Bp,
                                            int kt, int ntiles, int nt, int lane) {
  return *(const v16h*)(Bp + (((size_t)(kt * ntiles + nt) * 32 + lane) << 4));
}

__device__ __forceinline__ v8f wmma_acc(v16h a, v16h b, v8f c) {
  return __builtin_amdgcn_wmma_f32_16x16x32_f16(false, a, false, b, (short)0, c,
                                                false, false);
}

// Stage a 16-row x K-half A slab into LDS using the gfx1250 async-copy path
// (GLOBAL_LOAD_ASYNC_TO_LDS_B128, ASYNCcnt) when available; plain b128
// load + ds store otherwise. blockDim.x == 128, K % 64 == 0.
__device__ __forceinline__ void stage_a_slab(const _Float16* __restrict__ A,
                                             _Float16* sA, int row0, int K) {
  const int nvec = (16 * K) >> 3;  // number of 16B chunks
  for (int i = threadIdx.x; i < nvec; i += 128) {
    const int r = (i << 3) / K;
    const int c = (i << 3) % K;
    const _Float16* g = A + (size_t)(row0 + r) * K + c;
#if HAVE_ASYNC_LDS
    __builtin_amdgcn_global_load_async_to_lds_b128(
        (glb_v4i*)g, (lds_v4i*)(sA + (i << 3)), 0, 0);
#else
    *(v8h*)(sA + (i << 3)) = *(const v8h*)g;
#endif
  }
#if HAVE_ASYNC_LDS
  __builtin_amdgcn_s_wait_asynccnt(0);
#endif
  __syncthreads();
}

// ---------------------------------------------------------------------------
// Weight repack: row-major f32 (K x Nn) -> WMMA-B fragment layout f16.
// element i of lane L in tile (kt,nt):  k = kt*32 + 8*(L>>4) + (i&7) + 16*(i>>3),
//                                       n = nt*16 + (L&15)
// ---------------------------------------------------------------------------
__global__ void k_pack_b(const float* __restrict__ W, _Float16* __restrict__ out,
                         int K, int Nn) {
  int idx = blockIdx.x * blockDim.x + threadIdx.x;
  if (idx >= K * Nn) return;
  int i = idx & 15;
  int lane = (idx >> 4) & 31;
  int tile = idx >> 9;
  int ntiles = Nn >> 4;
  int kt = tile / ntiles, nt = tile - kt * ntiles;
  int k = kt * 32 + ((lane >> 4) << 3) + (i & 7) + ((i >> 3) << 4);
  int n = nt * 16 + (lane & 15);
  out[idx] = (_Float16)W[(size_t)k * Nn + n];
}

__global__ void k_concat_qkv(const float* __restrict__ Wq, const float* __restrict__ Wk,
                             const float* __restrict__ Wv, float* __restrict__ Wqkv) {
  int idx = blockIdx.x * blockDim.x + threadIdx.x;
  if (idx >= kD * kQKVN) return;
  int c = idx % kQKVN, dd = idx / kQKVN;
  const float* W = (c < 192) ? Wq : ((c < 384) ? Wk : Wv);
  Wqkv[idx] = W[dd * 192 + (c % 192)];
}

// ---------------------------------------------------------------------------
// Embedding + RMS-style norm: hn = h/max(||h||,eps) * sqrt(D) * (gamma+1)
// ---------------------------------------------------------------------------
__global__ void k_embed(const float* __restrict__ x, const float* __restrict__ We,
                        const float* __restrict__ be, const float* __restrict__ gamma,
                        float* __restrict__ hn32, _Float16* __restrict__ hn16) {
  int idx = blockIdx.x * blockDim.x + threadIdx.x;
  if (idx >= kM) return;
  float xv = x[idx];
  float h[kD];
  float ss = 0.f;
#pragma unroll
  for (int d = 0; d < kD; ++d) { h[d] = xv * We[d] + be[d]; ss += h[d] * h[d]; }
  float nrm = fmaxf(sqrtf(ss), 1e-12f);
  float sc = 8.0f / nrm;  // sqrt(64)
#pragma unroll
  for (int d = 0; d < kD; ++d) {
    float val = h[d] * sc * (gamma[d] + 1.0f);
    hn32[(size_t)idx * kD + d] = val;
    hn16[(size_t)idx * kD + d] = (_Float16)val;
  }
}

// ---------------------------------------------------------------------------
// GEMM 1: hn (16384 x 64) @ Wqkv (64 x 576) -> scatter into q/k/v (B,H,N,DH)
// grid (1024, 9), block 128 (4 waves, one 16x16 tile each)
// ---------------------------------------------------------------------------
__global__ void __launch_bounds__(128)
k_gemm_qkv(const _Float16* __restrict__ A, const _Float16* __restrict__ Bp,
           float* __restrict__ q, float* __restrict__ kk, float* __restrict__ vv) {
  const int lane = threadIdx.x & 31;
  const int wave = threadIdx.x >> 5;
  const int row0 = blockIdx.x * 16;
  const int nt = blockIdx.y * 4 + wave;
  v8f acc = {};
#pragma unroll
  for (int k0 = 0; k0 < kD; k0 += 32) {
    v16h a = load_a_frag(A, kD, row0, k0, lane);
    v16h b = load_b_frag(Bp, k0 >> 5, kQKVN / 16, nt, lane);
    acc = wmma_acc(a, b, acc);
  }
  const int hh = lane >> 4, colL = lane & 15;
  const int c = nt * 16 + colL;
  const int which = c / 192, cc = c % 192;
  const int h = cc >> 5, d = cc & 31;
  float* dst = (which == 0) ? q : ((which == 1) ? kk : vv);
#pragma unroll
  for (int j = 0; j < 8; ++j) {
    int row = row0 + j + 8 * hh;
    int b_ = row >> 9, n = row & 511;
    dst[((size_t)((b_ * kH + h) * kN + n)) * kDH + d] = acc[j];
  }
}

// ---------------------------------------------------------------------------
// Compression MLP layer 1: 6144x512 @ 512x512, bias+ReLU, f16 out.
// Block = 128 threads (4 waves). A slab (16x512 halfs, 16KB) staged in LDS
// via async-to-LDS; each wave computes 4 adjacent 16x16 N-tiles so one
// ds-sourced A fragment feeds 4 back-to-back WMMAs.
// grid (M/16, Nn/256)
// ---------------------------------------------------------------------------
__global__ void __launch_bounds__(128)
k_gemm_relu(const _Float16* __restrict__ A, const _Float16* __restrict__ Bp,
            const float* __restrict__ bias, _Float16* __restrict__ out,
            int M, int K, int Nn) {
  __shared__ _Float16 sA[16 * kCD];
  const int lane = threadIdx.x & 31;
  const int wave = threadIdx.x >> 5;
  const int row0 = blockIdx.x * 16;
  const int ntiles = Nn >> 4;
  const int nt0 = (blockIdx.y * 4 + wave) * 4;

  stage_a_slab(A, sA, row0, K);

  v8f acc0 = {}, acc1 = {}, acc2 = {}, acc3 = {};
  for (int k0 = 0; k0 < K; k0 += 32) {
    const int kt = k0 >> 5;
    v16h a = load_a_frag_lds(sA, K, k0, lane);
    acc0 = wmma_acc(a, load_b_frag(Bp, kt, ntiles, nt0 + 0, lane), acc0);
    acc1 = wmma_acc(a, load_b_frag(Bp, kt, ntiles, nt0 + 1, lane), acc1);
    acc2 = wmma_acc(a, load_b_frag(Bp, kt, ntiles, nt0 + 2, lane), acc2);
    acc3 = wmma_acc(a, load_b_frag(Bp, kt, ntiles, nt0 + 3, lane), acc3);
  }

  const int hh = lane >> 4, colL = lane & 15;
#pragma unroll
  for (int t = 0; t < 4; ++t) {
    v8f acc = (t == 0) ? acc0 : (t == 1) ? acc1 : (t == 2) ? acc2 : acc3;
    const int c = (nt0 + t) * 16 + colL;
    const float bb = bias[c];
#pragma unroll
    for (int j = 0; j < 8; ++j) {
      float val = acc[j] + bb;
      val = val > 0.f ? val : 0.f;
      out[(size_t)(row0 + j + 8 * hh) * Nn + c] = (_Float16)val;
    }
  }
}

// ---------------------------------------------------------------------------
// MLP layer 2: hidden (6144 x 512) @ W2 (512 x 32) + bias ->
// ck/cv (B*H, 33, 32) with slot 0 reserved for mem token. block = 64 (2 waves)
// ---------------------------------------------------------------------------
__global__ void __launch_bounds__(64)
k_gemm_ck(const _Float16* __restrict__ A, const _Float16* __restrict__ Bp,
          const float* __restrict__ bias, float* __restrict__ out) {
  const int lane = threadIdx.x & 31;
  const int nt = threadIdx.x >> 5;    // 0..1
  const int row0 = blockIdx.x * 16;
  v8f acc = {};
  for (int k0 = 0; k0 < kCD; k0 += 32) {
    v16h a = load_a_frag(A, kCD, row0, k0, lane);
    v16h b = load_b_frag(Bp, k0 >> 5, 2, nt, lane);
    acc = wmma_acc(a, b, acc);
  }
  const int hh = lane >> 4, colL = lane & 15;
  const int c = nt * 16 + colL;
  const float bb = bias[c];
#pragma unroll
  for (int j = 0; j < 8; ++j) {
    int row = row0 + j + 8 * hh;          // (b*H+h)*nC + chunk
    int bh = row >> 5, ch = row & 31;
    out[((size_t)bh * (knC + 1) + 1 + ch) * kDH + c] = acc[j] + bb;
  }
}

// ---------------------------------------------------------------------------
// GEMM: attn (16384 x 192) @ Wo (192 x 64) -> f32 (16384 x 64)
// ---------------------------------------------------------------------------
__global__ void __launch_bounds__(128)
k_gemm_wo(const _Float16* __restrict__ A, const _Float16* __restrict__ Bp,
          float* __restrict__ out) {
  const int lane = threadIdx.x & 31;
  const int nt = threadIdx.x >> 5;   // 0..3, Nn=64
  const int row0 = blockIdx.x * 16;
  v8f acc = {};
#pragma unroll
  for (int k0 = 0; k0 < kH * kDH; k0 += 32) {
    v16h a = load_a_frag(A, kH * kDH, row0, k0, lane);
    v16h b = load_b_frag(Bp, k0 >> 5, 4, nt, lane);
    acc = wmma_acc(a, b, acc);
  }
  const int hh = lane >> 4, colL = lane & 15;
  const int c = nt * 16 + colL;
#pragma unroll
  for (int j = 0; j < 8; ++j)
    out[(size_t)(row0 + j + 8 * hh) * kD + c] = acc[j];
}

// ---------------------------------------------------------------------------
// Build compression MLP input: kc_in[bh, chunk, p*32+d] = k + k_pos (f16)
// ---------------------------------------------------------------------------
__global__ void k_build_cin(const float* __restrict__ kk, const float* __restrict__ vv,
                            const float* __restrict__ k_pos, const float* __restrict__ v_pos,
                            _Float16* __restrict__ kc_in, _Float16* __restrict__ vc_in) {
  size_t idx = (size_t)blockIdx.x * blockDim.x + threadIdx.x;
  if (idx >= (size_t)kB * kH * kN * kDH) return;
  int d = idx & 31;
  size_t r = idx >> 5;
  int n = (int)(r & 511);
  size_t bh = r >> 9;
  int h = (int)(bh % kH);
  int p = n & 15, c = n >> 4;
  size_t row = bh * knC + c;
  size_t col = (size_t)p * kDH + d;
  kc_in[row * kCD + col] = (_Float16)(kk[idx] + k_pos[(h * kCB + p) * kDH + d]);
  vc_in[row * kCD + col] = (_Float16)(vv[idx] + v_pos[(h * kCB + p) * kDH + d]);
}

__global__ void k_memtok(const float* __restrict__ mem_k, const float* __restrict__ mem_v,
                         float* __restrict__ ck, float* __restrict__ cv) {
  int idx = blockIdx.x * blockDim.x + threadIdx.x;
  if (idx >= kB * kH * kDH) return;
  int d = idx & 31, bh = idx >> 5, h = bh % kH;
  ck[(size_t)bh * (knC + 1) * kDH + d] = mem_k[h * kDH + d];
  cv[(size_t)bh * (knC + 1) * kDH + d] = mem_v[h * kDH + d];
}

// ---------------------------------------------------------------------------
// RoPE: qr = rope(q) (new buffer), k = rope(k) in place (becomes kr)
// ---------------------------------------------------------------------------
__global__ void k_rope(const float* __restrict__ q, float* __restrict__ qr,
                       float* __restrict__ kk) {
  size_t idx = (size_t)blockIdx.x * blockDim.x + threadIdx.x;
  if (idx >= (size_t)kB * kH * kN * (kDH / 2)) return;
  int t = idx & 15;
  size_t r = idx >> 4;
  int n = (int)(r & 511);
  float inv = __expf(-((float)(2 * t) / 32.0f) * 9.210340371976184f); // 10000^-(2t/32)
  float fr = (float)n * inv;
  float cs = __cosf(fr), sn = __sinf(fr);
  size_t base = r * kDH + 2 * t;
  float q0 = q[base], q1 = q[base + 1];
  qr[base]     = q0 * cs - q1 * sn;
  qr[base + 1] = q1 * cs + q0 * sn;
  float k0 = kk[base], k1 = kk[base + 1];
  kk[base]     = k0 * cs - k1 * sn;
  kk[base + 1] = k1 * cs + k0 * sn;
}

// ---------------------------------------------------------------------------
// Attention: compressed + selected-fine + sliding + gate, per (b,h,n) thread.
// block = 128 threads, grid = B*H*(N/128); ck/cv staged in LDS.
// ---------------------------------------------------------------------------
__global__ void __launch_bounds__(128)
k_attn(const float* __restrict__ q, const float* __restrict__ qr,
       const float* __restrict__ kr, const float* __restrict__ vv,
       const float* __restrict__ ck, const float* __restrict__ cv,
       const float* __restrict__ hn, const float* __restrict__ Wg,
       const float* __restrict__ bg, _Float16* __restrict__ attn_out) {
  __shared__ float s_ck[(knC + 1) * kDH];
  __shared__ float s_cv[(knC + 1) * kDH];
  const int bh = blockIdx.x >> 2;
  const int nb = blockIdx.x & 3;
  const int b = bh / kH, h = bh % kH;
  for (int i = threadIdx.x; i < (knC + 1) * kDH; i += 128) {
    s_ck[i] = ck[(size_t)bh * (knC + 1) * kDH + i];
    s_cv[i] = cv[(size_t)bh * (knC + 1) * kDH + i];
  }
  __syncthreads();
  const int n = nb * 128 + threadIdx.x;
  const size_t rowoff = ((size_t)bh * kN + n) * kDH;
  const float NINF = -__builtin_huge_valf();

  float qv[kDH], qrv[kDH];
#pragma unroll
  for (int d = 0; d < kDH; ++d) { qv[d] = q[rowoff + d]; qrv[d] = qr[rowoff + d]; }

  // ---- compressed attention over 33 chunk tokens ----
  float csim[knC + 1];
  float mx = kNEG;
  for (int j = 0; j < knC + 1; ++j) {
    float dt = 0.f;
#pragma unroll
    for (int d = 0; d < kDH; ++d) dt += qv[d] * s_ck[j * kDH + d];
    float s = ((j * kCB - 1) < n) ? dt * kSCALE : kNEG;  // j=0 -> last=-1 (mem token)
    csim[j] = s;
    mx = fmaxf(mx, s);
  }
  float sum = 0.f;
  float ex[knC + 1];
  for (int j = 0; j < knC + 1; ++j) { ex[j] = __expf(csim[j] - mx); sum += ex[j]; }
  const float invc = 1.f / sum;
  float c_out[kDH];
#pragma unroll
  for (int d = 0; d < kDH; ++d) c_out[d] = 0.f;
  for (int j = 0; j < knC + 1; ++j) {
    float w = ex[j] * invc;
#pragma unroll
    for (int d = 0; d < kDH; ++d) c_out[d] += w * s_cv[j * kDH + d];
  }

  // ---- importance scores over fine blocks + top-2 select ----
  const int qb = n >> 3;
  float mxi = -1000.0f;  // leading sentinel column
  for (int f = 0; f < knF; ++f) {
    float t = (qb > f) ? csim[1 + (f >> 1)] : kNEG;
    mxi = fmaxf(mxi, t);
  }
  float sumi = __expf(-1000.0f - mxi);
  for (int f = 0; f < knF; ++f) {
    float t = (qb > f) ? csim[1 + (f >> 1)] : kNEG;
    sumi += __expf(t - mxi);
  }
  float b1v = NINF; int i1 = 0;
  for (int f = 0; f < knF; ++f) {
    float t = (qb > f) ? csim[1 + (f >> 1)] : kNEG;
    if (t > b1v) { b1v = t; i1 = f; }
  }
  float b2v = NINF; int i2 = 0;
  for (int f = 0; f < knF; ++f) {
    if (f == i1) continue;
    float t = (qb > f) ? csim[1 + (f >> 1)] : kNEG;
    if (t > b2v) { b2v = t; i2 = f; }
  }
  const float val1 = __expf(b1v - mxi) / sumi;
  const float val2 = __expf(b2v - mxi) / sumi;

  // ---- fine attention over 3 blocks (2 selected + own) of SB=8 ----
  const bool m0 = val1 > 1e-10f, m1 = val2 > 1e-10f;
  float fsim[3 * kSB];
  float mxf = kNEG;
#pragma unroll
  for (int t = 0; t < 3; ++t) {
    int blkb = (t == 0) ? i1 : ((t == 1) ? i2 : qb);
#pragma unroll
    for (int p = 0; p < kSB; ++p) {
      const float* krow = kr + ((size_t)bh * kN + blkb * kSB + p) * kDH;
      float dt = 0.f;
#pragma unroll
      for (int d = 0; d < kDH; ++d) dt += qrv[d] * krow[d];
      bool m = (t == 0) ? m0 : ((t == 1) ? m1 : (p <= (n & 7)));
      float s = m ? dt * kSCALE : kNEG;
      fsim[t * kSB + p] = s;
      mxf = fmaxf(mxf, s);
    }
  }
  float sumf = 0.f;
#pragma unroll
  for (int j = 0; j < 3 * kSB; ++j) { fsim[j] = __expf(fsim[j] - mxf); sumf += fsim[j]; }
  const float invf = 1.f / sumf;
  float f_out[kDH];
#pragma unroll
  for (int d = 0; d < kDH; ++d) f_out[d] = 0.f;
#pragma unroll
  for (int t = 0; t < 3; ++t) {
    int blkb = (t == 0) ? i1 : ((t == 1) ? i2 : qb);
#pragma unroll
    for (int p = 0; p < kSB; ++p) {
      const float* vrow = vv + ((size_t)bh * kN + blkb * kSB + p) * kDH;
      float w = fsim[t * kSB + p] * invf;
#pragma unroll
      for (int d = 0; d < kDH; ++d) f_out[d] += w * vrow[d];
    }
  }

  // ---- sliding window (WIN+1 = 8 keys) ----
  float ssim[8];
  float mxs = kNEG;
#pragma unroll
  for (int o = 0; o < 8; ++o) {
    int kidx = n - o;
    bool vld = kidx >= 0;
    if (kidx < 0) kidx = 0;
    const float* krow = kr + ((size_t)bh * kN + kidx) * kDH;
    float dt = 0.f;
#pragma unroll
    for (int d = 0; d < kDH; ++d) dt += qrv[d] * krow[d];
    float s = vld ? dt * kSCALE : kNEG;
    ssim[o] = s;
    mxs = fmaxf(mxs, s);
  }
  float sums = 0.f;
#pragma unroll
  for (int o = 0; o < 8; ++o) { ssim[o] = __expf(ssim[o] - mxs); sums += ssim[o]; }
  const float invs = 1.f / sums;
  float s_out[kDH];
#pragma unroll
  for (int d = 0; d < kDH; ++d) s_out[d] = 0.f;
#pragma unroll
  for (int o = 0; o < 8; ++o) {
    int kidx = n - o; if (kidx < 0) kidx = 0;
    const float* vrow = vv + ((size_t)bh * kN + kidx) * kDH;
    float w = ssim[o] * invs;
#pragma unroll
    for (int d = 0; d < kDH; ++d) s_out[d] += w * vrow[d];
  }

  // ---- sigmoid gate + combine ----
  const float* hnrow = hn + ((size_t)(b * kN + n)) * kD;
  float g[3];
#pragma unroll
  for (int jj = 0; jj < 3; ++jj) {
    float a = bg[h * 3 + jj];
    for (int dd = 0; dd < kD; ++dd) a += hnrow[dd] * Wg[dd * (3 * kH) + h * 3 + jj];
    g[jj] = 1.f / (1.f + __expf(-a));
  }
  _Float16* orow = attn_out + ((size_t)(b * kN + n)) * (kH * kDH) + h * kDH;
#pragma unroll
  for (int d = 0; d < kDH; ++d)
    orow[d] = (_Float16)(g[0] * c_out[d] + g[1] * f_out[d] + g[2] * s_out[d]);
}

// ---------------------------------------------------------------------------
// Mean pool over N, then 2-layer GELU head
// ---------------------------------------------------------------------------
__global__ void k_pool(const float* __restrict__ out_wo, float* __restrict__ pooled) {
  int b = blockIdx.x, c = threadIdx.x;  // 64 threads
  float s = 0.f;
  for (int n = 0; n < kN; ++n) s += out_wo[(size_t)(b * kN + n) * kD + c];
  pooled[b * kD + c] = s * (1.0f / (float)kN);
}

__global__ void k_head(const float* __restrict__ pooled, const float* __restrict__ Wh1,
                       const float* __restrict__ bh1, const float* __restrict__ Wh2,
                       const float* __restrict__ bh2, float* __restrict__ out) {
  __shared__ float hid[kHID];
  int b = blockIdx.x, t = threadIdx.x;  // 32 threads
  float a = bh1[t];
  for (int i = 0; i < kD; ++i) a += pooled[b * kD + i] * Wh1[i * kHID + t];
  hid[t] = 0.5f * a * (1.0f + erff(a * 0.70710678118654752f));  // exact gelu
  __syncthreads();
  if (t < kOUT) {
    float o = bh2[t];
    for (int i = 0; i < kHID; ++i) o += hid[i] * Wh2[i * kOUT + t];
    out[b * kOUT + t] = o;
  }
}

// ---------------------------------------------------------------------------
// Host orchestration
// ---------------------------------------------------------------------------
extern "C" void kernel_launch(void* const* d_in, const int* in_sizes, int n_in,
                              void* d_out, int out_size, void* d_ws, size_t ws_size,
                              hipStream_t stream) {
  const float* x     = (const float*)d_in[0];
  const float* We    = (const float*)d_in[1];
  const float* be    = (const float*)d_in[2];
  const float* gamma = (const float*)d_in[3];
  const float* Wq    = (const float*)d_in[4];
  const float* Wk    = (const float*)d_in[5];
  const float* Wv    = (const float*)d_in[6];
  const float* k_pos = (const float*)d_in[7];
  const float* v_pos = (const float*)d_in[8];
  const float* mem_k = (const float*)d_in[9];
  const float* mem_v = (const float*)d_in[10];
  const float* Wkc1  = (const float*)d_in[11];
  const float* bkc1  = (const float*)d_in[12];
  const float* Wkc2  = (const float*)d_in[13];
  const float* bkc2  = (const float*)d_in[14];
  const float* Wvc1  = (const float*)d_in[15];
  const float* bvc1  = (const float*)d_in[16];
  const float* Wvc2  = (const float*)d_in[17];
  const float* bvc2  = (const float*)d_in[18];
  const float* Wg    = (const float*)d_in[19];
  const float* bg    = (const float*)d_in[20];
  const float* Wo    = (const float*)d_in[21];
  const float* Wh1   = (const float*)d_in[22];
  const float* bh1   = (const float*)d_in[23];
  const float* Wh2   = (const float*)d_in[24];
  const float* bh2   = (const float*)d_in[25];
  float* out = (float*)d_out;

  char* base = (char*)d_ws;
  size_t off = 0;
  auto alloc = [&](size_t bytes) -> void* {
    void* p = base + off;
    off += (bytes + 255) & ~(size_t)255;
    return p;
  };

  float*    hn32   = (float*)alloc((size_t)kM * kD * 4);
  _Float16* hn16   = (_Float16*)alloc((size_t)kM * kD * 2);
  float*    Wqkv32 = (float*)alloc((size_t)kD * kQKVN * 4);
  _Float16* WqkvP  = (_Float16*)alloc((size_t)kD * kQKVN * 2);
  _Float16* Wkc1P  = (_Float16*)alloc((size_t)kCD * kCD * 2);
  _Float16* Wvc1P  = (_Float16*)alloc((size_t)kCD * kCD * 2);
  _Float16* Wkc2P  = (_Float16*)alloc((size_t)kCD * kDH * 2);
  _Float16* Wvc2P  = (_Float16*)alloc((size_t)kCD * kDH * 2);
  _Float16* WoP    = (_Float16*)alloc((size_t)kH * kDH * kD * 2);
  float*    dq     = (float*)alloc((size_t)kB * kH * kN * kDH * 4);
  float*    dk     = (float*)alloc((size_t)kB * kH * kN * kDH * 4);  // becomes kr
  float*    dv     = (float*)alloc((size_t)kB * kH * kN * kDH * 4);
  float*    dqr    = (float*)alloc((size_t)kB * kH * kN * kDH * 4);
  _Float16* kc_in  = (_Float16*)alloc((size_t)kMC * kCD * 2);
  _Float16* vc_in  = (_Float16*)alloc((size_t)kMC * kCD * 2);
  _Float16* khid   = (_Float16*)alloc((size_t)kMC * kCD * 2);
  _Float16* vhid   = (_Float16*)alloc((size_t)kMC * kCD * 2);
  float*    ckbuf  = (float*)alloc((size_t)kB * kH * (knC + 1) * kDH * 4);
  float*    cvbuf  = (float*)alloc((size_t)kB * kH * (knC + 1) * kDH * 4);
  _Float16* attn16 = (_Float16*)alloc((size_t)kM * kH * kDH * 2);
  float*    out_wo = (float*)alloc((size_t)kM * kD * 4);
  float*    pooled = (float*)alloc((size_t)kB * kD * 4);
  (void)ws_size; (void)in_sizes; (void)n_in; (void)out_size;

  // 1) embedding + norm
  k_embed<<<kM / 128, 128, 0, stream>>>(x, We, be, gamma, hn32, hn16);

  // 2) weight repacks (WMMA B-fragment layouts)
  k_concat_qkv<<<(kD * kQKVN + 255) / 256, 256, 0, stream>>>(Wq, Wk, Wv, Wqkv32);
  k_pack_b<<<(kD * kQKVN + 255) / 256, 256, 0, stream>>>(Wqkv32, WqkvP, kD, kQKVN);
  k_pack_b<<<(kCD * kCD + 255) / 256, 256, 0, stream>>>(Wkc1, Wkc1P, kCD, kCD);
  k_pack_b<<<(kCD * kCD + 255) / 256, 256, 0, stream>>>(Wvc1, Wvc1P, kCD, kCD);
  k_pack_b<<<(kCD * kDH + 255) / 256, 256, 0, stream>>>(Wkc2, Wkc2P, kCD, kDH);
  k_pack_b<<<(kCD * kDH + 255) / 256, 256, 0, stream>>>(Wvc2, Wvc2P, kCD, kDH);
  k_pack_b<<<(kH * kDH * kD + 255) / 256, 256, 0, stream>>>(Wo, WoP, kH * kDH, kD);

  // 3) QKV projection (WMMA)
  k_gemm_qkv<<<dim3(kM / 16, kQKVN / 64), 128, 0, stream>>>(hn16, WqkvP, dq, dk, dv);

  // 4) compression inputs (k+k_pos, v+v_pos), then RoPE (k in-place -> kr)
  {
    size_t tot = (size_t)kB * kH * kN * kDH;
    k_build_cin<<<(unsigned)((tot + 255) / 256), 256, 0, stream>>>(dk, dv, k_pos, v_pos,
                                                                  kc_in, vc_in);
  }
  {
    size_t tot = (size_t)kB * kH * kN * (kDH / 2);
    k_rope<<<(unsigned)((tot + 255) / 256), 256, 0, stream>>>(dq, dqr, dk);
  }

  // 5) compression MLPs (WMMA + async LDS staging): 6144x512 @ 512x512 relu,
  //    then @ 512x32
  k_gemm_relu<<<dim3(kMC / 16, kCD / 256), 128, 0, stream>>>(kc_in, Wkc1P, bkc1, khid,
                                                             kMC, kCD, kCD);
  k_gemm_ck<<<dim3(kMC / 16, 1), 64, 0, stream>>>(khid, Wkc2P, bkc2, ckbuf);
  k_gemm_relu<<<dim3(kMC / 16, kCD / 256), 128, 0, stream>>>(vc_in, Wvc1P, bvc1, vhid,
                                                             kMC, kCD, kCD);
  k_gemm_ck<<<dim3(kMC / 16, 1), 64, 0, stream>>>(vhid, Wvc2P, bvc2, cvbuf);
  k_memtok<<<(kB * kH * kDH + 255) / 256, 256, 0, stream>>>(mem_k, mem_v, ckbuf, cvbuf);

  // 6) attention (compressed + fine top-2 + sliding + gate)
  k_attn<<<kB * kH * (kN / 128), 128, 0, stream>>>(dq, dqr, dk, dv, ckbuf, cvbuf,
                                                   hn32, Wg, bg, attn16);

  // 7) output projection (WMMA), mean pool, GELU head
  k_gemm_wo<<<dim3(kM / 16, 1), 128, 0, stream>>>(attn16, WoP, out_wo);
  k_pool<<<kB, kD, 0, stream>>>(out_wo, pooled);
  k_head<<<kB, kHID, 0, stream>>>(pooled, Wh1, bh1, Wh2, bh2, out);
}
